// AttentionBlock_19189913879182
// MI455X (gfx1250) — matrix operand order repaired and checked by reading, not compiled, not run on hardware
//
#include <hip/hip_runtime.h>
#include <hip/hip_bf16.h>
#include <stdint.h>

// ---------------------------------------------------------------------------
// AttentionBlock for MI455X (gfx1250): GN -> QKV -> flash-attention -> proj
// bf16 WMMA (v_wmma_f32_16x16x32_bf16) for all GEMMs, f32 accumulation.
// Attention k chunk: TDM tensor_load_to_lds (TENSORcnt, HW row padding).
// Attention v chunk: GLOBAL_LOAD_ASYNC_TO_LDS_B128 (ASYNCcnt).
// Both double-buffered so the DMA engines overlap the WMMA pipeline.
// ---------------------------------------------------------------------------

typedef __bf16 bf16_t;
typedef __attribute__((ext_vector_type(16))) __bf16 v16bf;
typedef __attribute__((ext_vector_type(8)))  float  v8f;
typedef __attribute__((ext_vector_type(4)))  unsigned v4u;
typedef __attribute__((ext_vector_type(8)))  unsigned v8u;

#define BDIM 4
#define CDIM 256
#define NDIM 4096
#define GRP  32
#define CPG  8            // channels per group
#define ATT_SCALE 0.0625f // 256^-0.5
#define EPSV 1e-5f

union AF  { v16bf v; bf16_t e[16]; uint4 q[2]; };  // A or B fragment (16 bf16/lane)
union DF  { v8f   v; float  f[8]; };               // C/D fragment (8 f32/lane)
union PK8 { uint4 u; bf16_t e[8]; };               // 8 bf16 pack for 16B stores

__device__ __forceinline__ v8f wmma_bf16(v16bf a, v16bf b, v8f c) {
  // D = A(16x32 bf16) x B(32x16 bf16) + C(16x16 f32)
  return __builtin_amdgcn_wmma_f32_16x16x32_bf16(false, a, false, b,
                                                 (short)0, c, false, false);
}

// gfx1250 async memory->LDS copy (no VGPR data path, tracked by ASYNCcnt).
// LDS address operand = low 32 bits of the generic shared pointer
// (ISA 10.2).
__device__ __forceinline__ void async_b128(void* lds, const void* gsrc) {
  unsigned lo = (unsigned)(uintptr_t)lds;
  asm volatile("global_load_async_to_lds_b128 %0, %1, off"
               :: "v"(lo), "v"(gsrc) : "memory");
}
__device__ __forceinline__ void wait_async0() {
  asm volatile("s_wait_asynccnt 0x0" ::: "memory");
}

// gfx1250 Tensor Data Mover: 2D tile (rows x elems, 2B elements) from global
// to LDS with hardware row padding. Descriptor per ISA ch.8:
//  Group0: [1:0]=count=1, [63:32]=lds_addr, [120:64]=global_addr, [127:126]=2.
//  Group1: data_size=2B, pad_enable, pad_interval/amount, tensor/tile dims,
//          tensor_dim0_stride; workgroup_mask=0 (not in a cluster).
// pad_icode: interval = 2^(code+1) DWORDs ; pad_acode: amount = code+1 DWORDs.
__device__ __forceinline__ void tdm_load_2d(unsigned lds_addr, const void* gaddr,
                                            unsigned elems_per_row, unsigned rows,
                                            unsigned row_stride_elems,
                                            unsigned pad_icode, unsigned pad_acode) {
  unsigned long long ga = (unsigned long long)(uintptr_t)gaddr;
  v4u g0;
  g0[0] = 1u;                                  // count=1, user descriptor
  g0[1] = lds_addr;                            // lds_addr (bytes)
  g0[2] = (unsigned)ga;                        // global_addr[31:0]
  g0[3] = ((unsigned)(ga >> 32) & 0x01FFFFFFu) // global_addr[56:32]
        | (2u << 30);                          // type = 2 ("image")
  v8u g1;
  g1[0] = (1u << 16)                           // data_size: 2 bytes
        | (1u << 20)                           // pad_enable
        | (pad_icode << 22) | (pad_acode << 25);
  g1[1] = (elems_per_row & 0xFFFFu) << 16;     // tensor_dim0[15:0]
  g1[2] = (elems_per_row >> 16)                // tensor_dim0[31:16]
        | ((rows & 0xFFFFu) << 16);            // tensor_dim1[15:0]
  g1[3] = (rows >> 16)                         // tensor_dim1[31:16]
        | (elems_per_row << 16);               // tile_dim0
  g1[4] = rows & 0xFFFFu;                      // tile_dim1 (tile_dim2 = 0)
  g1[5] = row_stride_elems;                    // tensor_dim0_stride[31:0]
  g1[6] = 0u;                                  // stride[47:32], dim1_stride lo
  g1[7] = 0u;                                  // dim1_stride hi
  asm volatile("tensor_load_to_lds %0, %1" :: "s"(g0), "s"(g1) : "memory");
}

// ---------------------------------------------------------------------------
// Kernel 1: GroupNorm statistics. One block per (batch, group).
// (b*256 + g*8)*4096 == blockIdx.x*8*4096 since channels of a group are contiguous.
// ---------------------------------------------------------------------------
__global__ __launch_bounds__(256) void gn_stats_kernel(const float* __restrict__ x,
                                                       float* __restrict__ stats) {
  __shared__ float ssum[256];
  __shared__ float ssq[256];
  const int t = threadIdx.x;
  const float* p = x + (size_t)blockIdx.x * (CPG * NDIM);
  float s = 0.f, q = 0.f;
  for (int i = t; i < (CPG * NDIM) / 4; i += 256) {
    float4 v = ((const float4*)p)[i];
    s += v.x + v.y + v.z + v.w;
    q += v.x * v.x + v.y * v.y + v.z * v.z + v.w * v.w;
  }
  ssum[t] = s; ssq[t] = q;
  __syncthreads();
  for (int o = 128; o > 0; o >>= 1) {
    if (t < o) { ssum[t] += ssum[t + o]; ssq[t] += ssq[t + o]; }
    __syncthreads();
  }
  if (t == 0) {
    const float inv = 1.0f / (float)(CPG * NDIM);
    float mean = ssum[0] * inv;
    float var  = ssq[0] * inv - mean * mean;
    stats[blockIdx.x * 2 + 0] = mean;
    stats[blockIdx.x * 2 + 1] = rsqrtf(var + EPSV);
  }
}

// ---------------------------------------------------------------------------
// Kernel 2: fused GroupNorm-apply + QKV GEMM.
//   qkv[o][n] = sum_c qkv_w[o][c] * gn(x)[c][n] + qkv_b[o]
// Block tile: 128 (o) x 32 (n), K-steps of 32. 8 waves, each 16 o-rows x 32 n.
// Outputs: q,k token-major [B][N][C] bf16; v channel-major [B][C][N] bf16.
// ---------------------------------------------------------------------------
__global__ __launch_bounds__(256) void gn_qkv_kernel(
    const float* __restrict__ x, const float* __restrict__ gn_w,
    const float* __restrict__ gn_b, const float* __restrict__ qkv_w,
    const float* __restrict__ qkv_b, const float* __restrict__ stats,
    bf16_t* __restrict__ qws, bf16_t* __restrict__ kws, bf16_t* __restrict__ vws) {
  __shared__ __align__(16) bf16_t wlds[128][40]; // A tile (o x c), +8 pad
  __shared__ __align__(16) bf16_t hlds[32][40];  // B tile as [n][c], +8 pad
  const int t = threadIdx.x, lane = t & 31, w = t >> 5;
  const int half = lane >> 4, l15 = lane & 15;
  const int n0 = blockIdx.x * 32, o0 = blockIdx.y * 128, b = blockIdx.z;

  DF acc0, acc1;
#pragma unroll
  for (int r = 0; r < 8; r++) { acc0.f[r] = 0.f; acc1.f[r] = 0.f; }

  for (int kc = 0; kc < 8; kc++) {
    __syncthreads();
    // Load 128x32 weight tile (f32 -> bf16)
#pragma unroll
    for (int i = 0; i < 4; i++) {
      int idx = t + i * 256;
      int o = idx >> 3, seg = idx & 7;
      float4 wv = *(const float4*)(qkv_w + (size_t)(o0 + o) * CDIM + kc * 32 + seg * 4);
      wlds[o][seg * 4 + 0] = (bf16_t)wv.x;
      wlds[o][seg * 4 + 1] = (bf16_t)wv.y;
      wlds[o][seg * 4 + 2] = (bf16_t)wv.z;
      wlds[o][seg * 4 + 3] = (bf16_t)wv.w;
    }
    // Load 32x32 x tile, normalize, store transposed as hlds[n][c]
    {
      int c = t >> 3, seg = t & 7;
      int ch = kc * 32 + c;
      float mean = stats[(b * GRP + (ch >> 3)) * 2 + 0];
      float rstd = stats[(b * GRP + (ch >> 3)) * 2 + 1];
      float ga = gn_w[ch], be = gn_b[ch];
      float4 xv = *(const float4*)(x + ((size_t)(b * CDIM + ch)) * NDIM + n0 + seg * 4);
      hlds[seg * 4 + 0][c] = (bf16_t)((xv.x - mean) * rstd * ga + be);
      hlds[seg * 4 + 1][c] = (bf16_t)((xv.y - mean) * rstd * ga + be);
      hlds[seg * 4 + 2][c] = (bf16_t)((xv.z - mean) * rstd * ga + be);
      hlds[seg * 4 + 3][c] = (bf16_t)((xv.w - mean) * rstd * ga + be);
    }
    __syncthreads();
    AF a, b0, b1;
    a.q[0] = *(const uint4*)&wlds[w * 16 + l15][8 * half];
    a.q[1] = *(const uint4*)&wlds[w * 16 + l15][16 + 8 * half];
    b0.q[0] = *(const uint4*)&hlds[l15][8 * half];
    b0.q[1] = *(const uint4*)&hlds[l15][16 + 8 * half];
    b1.q[0] = *(const uint4*)&hlds[16 + l15][8 * half];
    b1.q[1] = *(const uint4*)&hlds[16 + l15][16 + 8 * half];
    acc0.v = wmma_bf16(a.v, b0.v, acc0.v);
    acc1.v = wmma_bf16(a.v, b1.v, acc1.v);
  }

  // Epilogue: D element (M = r + 8*half, N = l15) ; o = ob + r, contiguous over r.
  const int ob = o0 + w * 16 + 8 * half;
#pragma unroll
  for (int j = 0; j < 2; j++) {
    const DF& ac = j ? acc1 : acc0;
    const int n = n0 + j * 16 + l15;
    if (ob < 2 * CDIM) {           // q or k: token-major, channels contiguous
      PK8 pk;
#pragma unroll
      for (int r = 0; r < 8; r++) pk.e[r] = (bf16_t)(ac.f[r] + qkv_b[ob + r]);
      bf16_t* dst = (ob < CDIM)
          ? (qws + ((size_t)(b * NDIM) + n) * CDIM + ob)
          : (kws + ((size_t)(b * NDIM) + n) * CDIM + (ob - CDIM));
      *(uint4*)dst = pk.u;
    } else {                        // v: channel-major
#pragma unroll
      for (int r = 0; r < 8; r++)
        vws[((size_t)(b * CDIM) + (ob - 2 * CDIM + r)) * NDIM + n] =
            (bf16_t)(ac.f[r] + qkv_b[ob + r]);
    }
  }
}

// ---------------------------------------------------------------------------
// Kernel 3: flash attention. Block = 8 waves x 16 queries = 128 queries.
// Each wave keeps its 16x256 q tile resident in registers; block streams
// 32-key k/v chunks through double-buffered LDS (k via TDM, v via async).
// ---------------------------------------------------------------------------
__global__ __launch_bounds__(256, 1) void attn_kernel(
    const bf16_t* __restrict__ qws, const bf16_t* __restrict__ kws,
    const bf16_t* __restrict__ vws, bf16_t* __restrict__ ows) {
  __shared__ __align__(16) bf16_t kbuf[2][32][264];   // [m][c], +16B HW pad/row
  __shared__ __align__(16) bf16_t vbuf[2][256][40];   // [c][m], +16B pad/row
  __shared__ __align__(16) bf16_t pbuf[8][16][32];    // per-wave P re-layout
  const int t = threadIdx.x, lane = t & 31, w = t >> 5;
  const int half = lane >> 4, l15 = lane & 15;
  const int b = blockIdx.y;
  const int qbase = blockIdx.x * 128 + w * 16;
  const int NCH = NDIM / 32;

  const bf16_t* kbase = kws + (size_t)(b * NDIM) * CDIM;
  const bf16_t* vbase = vws + (size_t)(b * CDIM) * NDIM;

  // Fill one k/v chunk into LDS buffer `buf`.
  // k: single TDM 2D descriptor (32 rows x 512B, contiguous), wave 0 only
  //    (TDM ignores EXEC; one DMA per block, row pad 128DW->code6 / 4DW->code3).
  // v: per-lane async b128 (256 rows x 64B, stride N).
  auto issue_chunk = [&](int mcc, int buf) {
    if (w == 0) {
      tdm_load_2d((unsigned)(uintptr_t)&kbuf[buf][0][0],
                  kbase + (size_t)mcc * 32 * CDIM,
                  /*elems*/CDIM, /*rows*/32, /*stride*/CDIM,
                  /*pad_icode 128DW*/6u, /*pad_acode 4DW*/3u);
    }
    const bf16_t* vsrc = vbase + mcc * 32;
#pragma unroll
    for (int i = 0; i < 4; i++) {
      int idx = t + i * 256;
      int c = idx >> 2, seg = idx & 3;
      async_b128(&vbuf[buf][c][seg * 8], vsrc + (size_t)c * NDIM + seg * 8);
    }
  };

  // Resident q fragments: A-layout, 8 K-steps of 32 channels.
  AF qf[8];
  {
    const bf16_t* qrow = qws + ((size_t)(b * NDIM) + qbase + l15) * CDIM;
#pragma unroll
    for (int s = 0; s < 8; s++) {
      qf[s].q[0] = *(const uint4*)(qrow + s * 32 + 8 * half);
      qf[s].q[1] = *(const uint4*)(qrow + s * 32 + 16 + 8 * half);
    }
  }

  DF acc[16];
#pragma unroll
  for (int ct = 0; ct < 16; ct++)
#pragma unroll
    for (int r = 0; r < 8; r++) acc[ct].f[r] = 0.f;
  float mrow[8], lrow[8];
#pragma unroll
  for (int r = 0; r < 8; r++) { mrow[r] = -3.0e38f; lrow[r] = 0.f; }

  issue_chunk(0, 0);                    // prologue: fill buffer 0

  for (int mc = 0; mc < NCH; mc++) {
    const int cur = mc & 1;
    __builtin_amdgcn_s_wait_tensorcnt(0);  // own TDM for chunk mc landed
    wait_async0();                         // own async copies landed
    __syncthreads();                       // all waves' copies landed; prev buf free
    if (mc + 1 < NCH) issue_chunk(mc + 1, cur ^ 1);  // overlap with compute

    // Scores: S(16q x 32keys) as two D-tiles, accumulate over 8 channel steps.
    DF s0, s1;
#pragma unroll
    for (int r = 0; r < 8; r++) { s0.f[r] = 0.f; s1.f[r] = 0.f; }
#pragma unroll
    for (int s = 0; s < 8; s++) {
      AF k0, k1;
      k0.q[0] = *(const uint4*)&kbuf[cur][l15][s * 32 + 8 * half];
      k0.q[1] = *(const uint4*)&kbuf[cur][l15][s * 32 + 16 + 8 * half];
      k1.q[0] = *(const uint4*)&kbuf[cur][16 + l15][s * 32 + 8 * half];
      k1.q[1] = *(const uint4*)&kbuf[cur][16 + l15][s * 32 + 16 + 8 * half];
      s0.v = wmma_bf16(qf[s].v, k0.v, s0.v);
      s1.v = wmma_bf16(qf[s].v, k1.v, s1.v);
    }

    // Online softmax. D row M = r + 8*half lives in a 16-lane half-group.
    float alpha[8];
#pragma unroll
    for (int r = 0; r < 8; r++) {
      float a = s0.f[r] * ATT_SCALE, c = s1.f[r] * ATT_SCALE;
      float mx = fmaxf(a, c);
#pragma unroll
      for (int msk = 1; msk < 16; msk <<= 1) mx = fmaxf(mx, __shfl_xor(mx, msk, 32));
      float mn = fmaxf(mrow[r], mx);
      alpha[r] = __expf(mrow[r] - mn);
      mrow[r] = mn;
      float p0 = __expf(a - mn), p1 = __expf(c - mn);
      s0.f[r] = p0; s1.f[r] = p1;
      float rs = p0 + p1;
#pragma unroll
      for (int msk = 1; msk < 16; msk <<= 1) rs += __shfl_xor(rs, msk, 32);
      lrow[r] = lrow[r] * alpha[r] + rs;
    }
#pragma unroll
    for (int ct = 0; ct < 16; ct++)
#pragma unroll
      for (int r = 0; r < 8; r++) acc[ct].f[r] *= alpha[r];

    // Re-layout P (D-layout) -> A-fragment via per-wave LDS bounce.
#pragma unroll
    for (int r = 0; r < 8; r++) {
      pbuf[w][r + 8 * half][l15]      = (bf16_t)s0.f[r];
      pbuf[w][r + 8 * half][16 + l15] = (bf16_t)s1.f[r];
    }
    AF pf;
    pf.q[0] = *(const uint4*)&pbuf[w][l15][8 * half];
    pf.q[1] = *(const uint4*)&pbuf[w][l15][16 + 8 * half];

    // out += P(16x32) x V^T(32x16) over 16 channel tiles
#pragma unroll
    for (int ct = 0; ct < 16; ct++) {
      AF vf;
      vf.q[0] = *(const uint4*)&vbuf[cur][ct * 16 + l15][8 * half];
      vf.q[1] = *(const uint4*)&vbuf[cur][ct * 16 + l15][16 + 8 * half];
      acc[ct].v = wmma_bf16(pf.v, vf.v, acc[ct].v);
    }
  }

  // Finalize: out[c][n] channel-major bf16; 8 contiguous tokens per lane.
  float inv[8];
#pragma unroll
  for (int r = 0; r < 8; r++) inv[r] = 1.0f / lrow[r];
#pragma unroll
  for (int ct = 0; ct < 16; ct++) {
    PK8 pk;
#pragma unroll
    for (int r = 0; r < 8; r++) pk.e[r] = (bf16_t)(acc[ct].f[r] * inv[r]);
    int c = ct * 16 + l15;
    *(uint4*)(ows + ((size_t)(b * CDIM) + c) * NDIM + qbase + 8 * half) = pk.u;
  }
}

// ---------------------------------------------------------------------------
// Kernel 4: proj GEMM + bias + residual. y = proj_w @ attn_out + proj_b + x.
// Block tile 128(o) x 32(n); output f32 straight into [B][C][H][W].
// ---------------------------------------------------------------------------
__global__ __launch_bounds__(256) void proj_kernel(
    const float* __restrict__ x, const float* __restrict__ proj_w,
    const float* __restrict__ proj_b, const bf16_t* __restrict__ ows,
    float* __restrict__ out) {
  __shared__ __align__(16) bf16_t wlds[128][40];
  __shared__ __align__(16) bf16_t hlds[32][40];
  const int t = threadIdx.x, lane = t & 31, w = t >> 5;
  const int half = lane >> 4, l15 = lane & 15;
  const int n0 = blockIdx.x * 32, o0 = blockIdx.y * 128, b = blockIdx.z;

  DF acc0, acc1;
#pragma unroll
  for (int r = 0; r < 8; r++) { acc0.f[r] = 0.f; acc1.f[r] = 0.f; }

  for (int kc = 0; kc < 8; kc++) {
    __syncthreads();
#pragma unroll
    for (int i = 0; i < 4; i++) {
      int idx = t + i * 256;
      int o = idx >> 3, seg = idx & 7;
      float4 wv = *(const float4*)(proj_w + (size_t)(o0 + o) * CDIM + kc * 32 + seg * 4);
      wlds[o][seg * 4 + 0] = (bf16_t)wv.x;
      wlds[o][seg * 4 + 1] = (bf16_t)wv.y;
      wlds[o][seg * 4 + 2] = (bf16_t)wv.z;
      wlds[o][seg * 4 + 3] = (bf16_t)wv.w;
    }
    {
      int c = t >> 3, seg = t & 7;
      const bf16_t* src = ows + ((size_t)(b * CDIM) + kc * 32 + c) * NDIM + n0 + seg * 4;
#pragma unroll
      for (int jj = 0; jj < 4; jj++) hlds[seg * 4 + jj][c] = src[jj];
    }
    __syncthreads();
    AF a, b0, b1;
    a.q[0] = *(const uint4*)&wlds[w * 16 + l15][8 * half];
    a.q[1] = *(const uint4*)&wlds[w * 16 + l15][16 + 8 * half];
    b0.q[0] = *(const uint4*)&hlds[l15][8 * half];
    b0.q[1] = *(const uint4*)&hlds[l15][16 + 8 * half];
    b1.q[0] = *(const uint4*)&hlds[16 + l15][8 * half];
    b1.q[1] = *(const uint4*)&hlds[16 + l15][16 + 8 * half];
    acc0.v = wmma_bf16(a.v, b0.v, acc0.v);
    acc1.v = wmma_bf16(a.v, b1.v, acc1.v);
  }

  const int ob = o0 + w * 16 + 8 * half;
#pragma unroll
  for (int j = 0; j < 2; j++) {
    const DF& ac = j ? acc1 : acc0;
    const int n = n0 + j * 16 + l15;
#pragma unroll
    for (int r = 0; r < 8; r++) {
      size_t addr = ((size_t)(b * CDIM) + ob + r) * NDIM + n;
      out[addr] = ac.f[r] + proj_b[ob + r] + x[addr];
    }
  }
}

// ---------------------------------------------------------------------------
extern "C" void kernel_launch(void* const* d_in, const int* in_sizes, int n_in,
                              void* d_out, int out_size, void* d_ws, size_t ws_size,
                              hipStream_t stream) {
  const float* x      = (const float*)d_in[0];
  const float* gn_w   = (const float*)d_in[1];
  const float* gn_b   = (const float*)d_in[2];
  const float* qkv_w  = (const float*)d_in[3];
  const float* qkv_b  = (const float*)d_in[4];
  const float* proj_w = (const float*)d_in[5];
  const float* proj_b = (const float*)d_in[6];
  float* out = (float*)d_out;

  char* ws = (char*)d_ws;
  float* stats = (float*)ws;                         // 128 * 2 f32
  size_t off = 4096;
  const size_t plane = (size_t)BDIM * NDIM * CDIM * sizeof(bf16_t); // 8 MB
  bf16_t* qws = (bf16_t*)(ws + off); off += plane;   // [B][N][C]
  bf16_t* kws = (bf16_t*)(ws + off); off += plane;   // [B][N][C]
  bf16_t* vws = (bf16_t*)(ws + off); off += plane;   // [B][C][N]
  bf16_t* ows = (bf16_t*)(ws + off);                 // [B][C][N]

  gn_stats_kernel<<<dim3(BDIM * GRP), dim3(256), 0, stream>>>(x, stats);
  gn_qkv_kernel<<<dim3(NDIM / 32, (3 * CDIM) / 128, BDIM), dim3(256), 0, stream>>>(
      x, gn_w, gn_b, qkv_w, qkv_b, stats, qws, kws, vws);
  attn_kernel<<<dim3(NDIM / 128, BDIM), dim3(256), 0, stream>>>(qws, kws, vws, ows);
  proj_kernel<<<dim3(NDIM / 32, CDIM / 128, BDIM), dim3(256), 0, stream>>>(
      x, proj_w, proj_b, ows, out);
}
